// Moe_64381559767963
// MI455X (gfx1250) — matrix operand order repaired and checked by reading, not compiled, not run on hardware
//
#include <hip/hip_runtime.h>

// ---------------------------------------------------------------------------
// MoE (DeepSeek-style) for MI455X / gfx1250, wave32, f16 WMMA w/ f32 accum.
// Round 2: + async global->LDS (ASYNCcnt) for f16 A tiles, + global_prefetch
// of next-K weight rows.
// ---------------------------------------------------------------------------

#define DIM       4096
#define INTER     1024
#define NEXP      16
#define NGROUPS   4
#define TOKENS    2048
#define SH_INTER  2048
#define TM        128   // M tile
#define TN        128   // N tile
#define TK        32    // K step
#define LDSROW    40    // padded row stride in _Float16 elems (80B) to avoid bank conflicts

typedef __attribute__((ext_vector_type(16))) _Float16 v16h;
typedef __attribute__((ext_vector_type(8)))  float    v8f;

// ---- helpers ---------------------------------------------------------------

__device__ inline uint4 cvt8_f16(float4 a, float4 b) {
    union { _Float16 h[8]; uint4 q; } p;
    p.h[0] = (_Float16)a.x; p.h[1] = (_Float16)a.y;
    p.h[2] = (_Float16)a.z; p.h[3] = (_Float16)a.w;
    p.h[4] = (_Float16)b.x; p.h[5] = (_Float16)b.y;
    p.h[6] = (_Float16)b.z; p.h[7] = (_Float16)b.w;
    return p.q;
}

__device__ inline unsigned pkh(float a, float b) {
    union { _Float16 h[2]; unsigned u; } p;
    p.h[0] = (_Float16)a; p.h[1] = (_Float16)b;
    return p.u;
}

// LDS byte offset of a __shared__ pointer: low 32 bits of the generic address
// (AS3->AS0 cast places the offset in addr[31:0] per the aperture scheme).
__device__ inline unsigned lds_off_of(const void* p) {
    return (unsigned)(uintptr_t)p;
}

// A fragment: lane 0-15 -> M=lane, K {8h..8h+7, 16+8h..16+8h+7}; h = lane>>4
__device__ inline v16h load_af(const _Float16* s, int row, int half) {
    union { v16h v; uint4 q[2]; } f;
    f.q[0] = *(const uint4*)&s[row * LDSROW + 8 * half];
    f.q[1] = *(const uint4*)&s[row * LDSROW + 16 + 8 * half];
    return f.v;
}

// B fragment: lane 0-15 -> N=lane, K {8h..8h+7, 16+8h..16+8h+7}; h = lane>>4 (B^T in LDS)
__device__ inline v16h load_bf(const _Float16* s, int col, int half) {
    union { v16h v; uint4 q[2]; } f;
    f.q[0] = *(const uint4*)&s[col * LDSROW + 8 * half];
    f.q[1] = *(const uint4*)&s[col * LDSROW + 16 + 8 * half];
    return f.v;
}

__device__ inline void storeB(_Float16* sB, int nb, int kp,
                              float4 a0, float4 a1, float4 b0, float4 b1) {
    // rows k=2kp (a0,a1) and k=2kp+1 (b0,b1), cols nb..nb+7; LDS holds B^T [n][k]
    *(unsigned*)&sB[(nb + 0) * LDSROW + 2 * kp] = pkh(a0.x, b0.x);
    *(unsigned*)&sB[(nb + 1) * LDSROW + 2 * kp] = pkh(a0.y, b0.y);
    *(unsigned*)&sB[(nb + 2) * LDSROW + 2 * kp] = pkh(a0.z, b0.z);
    *(unsigned*)&sB[(nb + 3) * LDSROW + 2 * kp] = pkh(a0.w, b0.w);
    *(unsigned*)&sB[(nb + 4) * LDSROW + 2 * kp] = pkh(a1.x, b1.x);
    *(unsigned*)&sB[(nb + 5) * LDSROW + 2 * kp] = pkh(a1.y, b1.y);
    *(unsigned*)&sB[(nb + 6) * LDSROW + 2 * kp] = pkh(a1.z, b1.z);
    *(unsigned*)&sB[(nb + 7) * LDSROW + 2 * kp] = pkh(a1.w, b1.w);
}

#define WMMA_F16(A, B, C) \
    __builtin_amdgcn_wmma_f32_16x16x32_f16(false, (A), false, (B), (short)0, (C), false, false)

// ---- 1) router: logits -> softmax -> +bias -> group-limited top-2 ----------

__global__ __launch_bounds__(128) void router_kernel(
    const float* __restrict__ X, const float* __restrict__ RW,
    const float* __restrict__ RB, const float* __restrict__ Rbias,
    int* __restrict__ topidx, float* __restrict__ topval)
{
    const int t    = blockIdx.x * 4 + (threadIdx.x >> 5);
    const int lane = threadIdx.x & 31;

    float acc[NEXP];
#pragma unroll
    for (int e = 0; e < NEXP; ++e) acc[e] = 0.f;

    const float* xr = X + (size_t)t * DIM;
    for (int d = lane; d < DIM; d += 32) {
        float xv = xr[d];
        const float4* w = (const float4*)(RW + (size_t)d * NEXP);
        float4 w0 = w[0], w1 = w[1], w2 = w[2], w3 = w[3];
        acc[0]  = fmaf(xv, w0.x, acc[0]);  acc[1]  = fmaf(xv, w0.y, acc[1]);
        acc[2]  = fmaf(xv, w0.z, acc[2]);  acc[3]  = fmaf(xv, w0.w, acc[3]);
        acc[4]  = fmaf(xv, w1.x, acc[4]);  acc[5]  = fmaf(xv, w1.y, acc[5]);
        acc[6]  = fmaf(xv, w1.z, acc[6]);  acc[7]  = fmaf(xv, w1.w, acc[7]);
        acc[8]  = fmaf(xv, w2.x, acc[8]);  acc[9]  = fmaf(xv, w2.y, acc[9]);
        acc[10] = fmaf(xv, w2.z, acc[10]); acc[11] = fmaf(xv, w2.w, acc[11]);
        acc[12] = fmaf(xv, w3.x, acc[12]); acc[13] = fmaf(xv, w3.y, acc[13]);
        acc[14] = fmaf(xv, w3.z, acc[14]); acc[15] = fmaf(xv, w3.w, acc[15]);
    }
#pragma unroll
    for (int off = 16; off > 0; off >>= 1)
#pragma unroll
        for (int e = 0; e < NEXP; ++e)
            acc[e] += __shfl_xor(acc[e], off, 32);

    float logit[NEXP];
    float mx = -1e30f;
#pragma unroll
    for (int e = 0; e < NEXP; ++e) { logit[e] = acc[e] + RB[e]; mx = fmaxf(mx, logit[e]); }
    float sum = 0.f, sc[NEXP];
#pragma unroll
    for (int e = 0; e < NEXP; ++e) { sc[e] = __expf(logit[e] - mx); sum += sc[e]; }
    float inv = 1.f / sum;
#pragma unroll
    for (int e = 0; e < NEXP; ++e) sc[e] = sc[e] * inv + Rbias[e];

    float gmax[NGROUPS];
#pragma unroll
    for (int g = 0; g < NGROUPS; ++g) {
        float m0 = fmaxf(sc[4 * g + 0], sc[4 * g + 1]);
        float m1 = fmaxf(sc[4 * g + 2], sc[4 * g + 3]);
        gmax[g] = fmaxf(m0, m1);
    }
    int g0 = 0;
#pragma unroll
    for (int g = 1; g < NGROUPS; ++g) if (gmax[g] > gmax[g0]) g0 = g;
    int g1 = (g0 == 0) ? 1 : 0;
#pragma unroll
    for (int g = 0; g < NGROUPS; ++g)
        if (g != g0 && gmax[g] > gmax[g1]) g1 = g;

    int i0 = -1; float v0 = -1e30f;
#pragma unroll
    for (int e = 0; e < NEXP; ++e) {
        bool ok = ((e >> 2) == g0) || ((e >> 2) == g1);
        if (ok && sc[e] > v0) { v0 = sc[e]; i0 = e; }
    }
    int i1 = -1; float v1 = -1e30f;
#pragma unroll
    for (int e = 0; e < NEXP; ++e) {
        bool ok = (((e >> 2) == g0) || ((e >> 2) == g1)) && (e != i0);
        if (ok && sc[e] > v1) { v1 = sc[e]; i1 = e; }
    }
    if (lane == 0) {
        topidx[t * 2 + 0] = i0; topidx[t * 2 + 1] = i1;
        topval[t * 2 + 0] = v0; topval[t * 2 + 1] = v1;   // moe_route_scale = 1.0
    }
}

// ---- 2) build compact per-expert token lists (deterministic, single block) -

__global__ __launch_bounds__(512) void build_lists(
    const int* __restrict__ topidx, int* __restrict__ toklist,
    int* __restrict__ offsets)
{
    __shared__ int cnts[NEXP];
    __shared__ int pref[NEXP + 1];
    const int wid = threadIdx.x >> 5;   // one wave per expert
    const int lane = threadIdx.x & 31;

    int c = 0;
    for (int t = lane; t < TOKENS; t += 32) {
        if (topidx[t * 2] == wid || topidx[t * 2 + 1] == wid) c++;
    }
#pragma unroll
    for (int off = 16; off > 0; off >>= 1) c += __shfl_xor(c, off, 32);
    if (lane == 0) cnts[wid] = c;
    __syncthreads();
    if (threadIdx.x == 0) {
        int s = 0;
        for (int e = 0; e < NEXP; ++e) { pref[e] = s; offsets[e] = s; s += cnts[e]; }
        pref[NEXP] = s; offsets[NEXP] = s;
    }
    __syncthreads();

    int base = pref[wid];
    for (int t0 = 0; t0 < TOKENS; t0 += 32) {
        int t = t0 + lane;
        int slot = -1;
        if (topidx[t * 2] == wid) slot = 0;
        else if (topidx[t * 2 + 1] == wid) slot = 1;
        unsigned m = (unsigned)__ballot(slot >= 0);
        if (slot >= 0) {
            int rank = __popc(m & ((1u << lane) - 1u));
            toklist[base + rank] = t | (slot << 16);
        }
        base += __popc(m);
    }
}

// ---- 3) dual GEMM + SiLU:  H = silu(Xg@W1+b1) * (Xg@W2+b2)  -> f16 buffer --

__global__ __launch_bounds__(256) void fused_gemm_silu(
    const float* __restrict__ X, const int* __restrict__ toklist,
    const int* __restrict__ offsets, int Mfix,
    const float* __restrict__ W1, const float* __restrict__ Bias1,
    const float* __restrict__ W2, const float* __restrict__ Bias2,
    int Kdim, int Ndim, _Float16* __restrict__ Hout)
{
    __shared__ __align__(16) _Float16 sA [TM * LDSROW];
    __shared__ __align__(16) _Float16 sB1[TN * LDSROW];
    __shared__ __align__(16) _Float16 sB2[TN * LDSROW];

    const int e = blockIdx.z;
    int off = 0, cnt = Mfix;
    if (offsets) { off = offsets[e]; cnt = offsets[e + 1] - off; }
    const int m0 = blockIdx.y * TM;
    if (m0 >= cnt) return;
    const int n0 = blockIdx.x * TN;

    const size_t ws = (size_t)Kdim * Ndim;
    const float* w1 = W1 + (size_t)e * ws;
    const float* w2 = W2 + (size_t)e * ws;
    const float* b1 = Bias1 + (size_t)e * Ndim;
    const float* b2 = Bias2 + (size_t)e * Ndim;

    const int tid  = threadIdx.x;
    const int lane = tid & 31, wid = tid >> 5;
    const int half = lane >> 4, l15 = lane & 15;
    const int wm = (wid & 3) * 32, wn = (wid >> 2) * 64;

    const int ar = tid >> 1;
    const int ah = (tid & 1) * 16;
    int mi = m0 + ar; if (mi >= cnt) mi = cnt - 1;
    const int arow = toklist ? (toklist[off + mi] & 0xffff) : mi;
    const float* aptr = X + (size_t)arow * Kdim + ah;

    const int kp = tid >> 4;
    const int nb = (tid & 15) * 8;
    const float* w1p = w1 + (size_t)(2 * kp) * Ndim + n0 + nb;
    const float* w2p = w2 + (size_t)(2 * kp) * Ndim + n0 + nb;

    v8f acc1[2][4] = {};
    v8f acc2[2][4] = {};

    for (int k0 = 0; k0 < Kdim; k0 += TK) {
        float4 a0 = *(const float4*)(aptr + k0 + 0);
        float4 a1 = *(const float4*)(aptr + k0 + 4);
        float4 a2 = *(const float4*)(aptr + k0 + 8);
        float4 a3 = *(const float4*)(aptr + k0 + 12);
        const float* p1 = w1p + (size_t)k0 * Ndim;
        float4 q10 = *(const float4*)(p1);
        float4 q11 = *(const float4*)(p1 + 4);
        float4 q12 = *(const float4*)(p1 + Ndim);
        float4 q13 = *(const float4*)(p1 + Ndim + 4);
        const float* p2 = w2p + (size_t)k0 * Ndim;
        float4 q20 = *(const float4*)(p2);
        float4 q21 = *(const float4*)(p2 + 4);
        float4 q22 = *(const float4*)(p2 + Ndim);
        float4 q23 = *(const float4*)(p2 + Ndim + 4);

        // prefetch next K-step's weight rows (streams once; hide HBM latency)
        if (k0 + TK < Kdim) {
            __builtin_prefetch(p1 + (size_t)TK * Ndim, 0, 1);
            __builtin_prefetch(p2 + (size_t)TK * Ndim, 0, 1);
        }

        __syncthreads();
        *(uint4*)&sA[ar * LDSROW + ah]     = cvt8_f16(a0, a1);
        *(uint4*)&sA[ar * LDSROW + ah + 8] = cvt8_f16(a2, a3);
        storeB(sB1, nb, kp, q10, q11, q12, q13);
        storeB(sB2, nb, kp, q20, q21, q22, q23);
        __syncthreads();

        v16h af[2];
#pragma unroll
        for (int im = 0; im < 2; ++im)
            af[im] = load_af(sA, wm + im * 16 + l15, half);
#pragma unroll
        for (int in = 0; in < 4; ++in) {
            v16h bf1 = load_bf(sB1, wn + in * 16 + l15, half);
            v16h bf2 = load_bf(sB2, wn + in * 16 + l15, half);
#pragma unroll
            for (int im = 0; im < 2; ++im) {
                acc1[im][in] = WMMA_F16(af[im], bf1, acc1[im][in]);
                acc2[im][in] = WMMA_F16(af[im], bf2, acc2[im][in]);
            }
        }
    }

#pragma unroll
    for (int im = 0; im < 2; ++im)
#pragma unroll
        for (int in = 0; in < 4; ++in) {
            const int n = n0 + wn + in * 16 + l15;
            const float bb1 = b1[n], bb2 = b2[n];
#pragma unroll
            for (int v = 0; v < 8; ++v) {
                const int mloc = wm + im * 16 + v + 8 * half;
                if (m0 + mloc < cnt) {
                    float h1 = acc1[im][in][v] + bb1;
                    float h2 = acc2[im][in][v] + bb2;
                    float hh = (h1 / (1.f + __expf(-h1))) * h2;
                    Hout[(size_t)(off + m0 + mloc) * Ndim + n] = (_Float16)hh;
                }
            }
        }
}

// ---- 4) output GEMM:  Y = H@W3 + b3, gated scatter (routed) or plain store -
// A (activations) is already f16: stream it straight into LDS with the CDNA5
// async-copy path (ASYNCcnt), bypassing VGPR staging.

__global__ __launch_bounds__(256) void gemm_out(
    const _Float16* __restrict__ Abuf, const int* __restrict__ toklist,
    const int* __restrict__ offsets, int Mfix,
    const float* __restrict__ W3, const float* __restrict__ Bias3,
    const float* __restrict__ gatev,
    int Kdim, int Ndim, float* __restrict__ Yout)
{
    __shared__ __align__(16) _Float16 sA[TM * LDSROW];
    __shared__ __align__(16) _Float16 sB[TN * LDSROW];

    const int e = blockIdx.z;
    int off = 0, cnt = Mfix;
    if (offsets) { off = offsets[e]; cnt = offsets[e + 1] - off; }
    const int m0 = blockIdx.y * TM;
    if (m0 >= cnt) return;
    const int n0 = blockIdx.x * TN;

    const float* w3 = W3 + (size_t)e * Kdim * Ndim;
    const float* b3 = Bias3 + (size_t)e * Ndim;

    const int tid  = threadIdx.x;
    const int lane = tid & 31, wid = tid >> 5;
    const int half = lane >> 4, l15 = lane & 15;
    const int wm = (wid & 3) * 32, wn = (wid >> 2) * 64;

    const int ar = tid >> 1;
    const int ah = (tid & 1) * 16;
    int mi = m0 + ar; if (mi >= cnt) mi = cnt - 1;
    const _Float16* aptr = Abuf + (size_t)(off + mi) * Kdim + ah;

    const int kp = tid >> 4;
    const int nb = (tid & 15) * 8;
    const float* wp = w3 + (size_t)(2 * kp) * Ndim + n0 + nb;

    // per-thread LDS destinations for the async A copies (two 16B chunks)
    const unsigned ldsA0 = lds_off_of(&sA[ar * LDSROW + ah]);
    const unsigned ldsA1 = lds_off_of(&sA[ar * LDSROW + ah + 8]);

    v8f acc[2][4] = {};

    for (int k0 = 0; k0 < Kdim; k0 += TK) {
        const float* p = wp + (size_t)k0 * Ndim;
        float4 q0 = *(const float4*)(p);
        float4 q1 = *(const float4*)(p + 4);
        float4 q2 = *(const float4*)(p + Ndim);
        float4 q3 = *(const float4*)(p + Ndim + 4);

        if (k0 + TK < Kdim)
            __builtin_prefetch(p + (size_t)TK * Ndim, 0, 1);

        __syncthreads();   // all waves done reading previous LDS tiles

        // async global->LDS copy of this thread's 32B slice of the A tile
        {
            const _Float16* g0 = aptr + k0;
            const _Float16* g1 = aptr + k0 + 8;
            asm volatile(
                "global_load_async_to_lds_b128 %0, %2, off\n\t"
                "global_load_async_to_lds_b128 %1, %3, off"
                :
                : "v"(ldsA0), "v"(ldsA1), "v"(g0), "v"(g1)
                : "memory");
        }
        storeB(sB, nb, kp, q0, q1, q2, q3);
        asm volatile("s_wait_asynccnt 0x0" ::: "memory");
        __syncthreads();

        v16h af[2];
#pragma unroll
        for (int im = 0; im < 2; ++im)
            af[im] = load_af(sA, wm + im * 16 + l15, half);
#pragma unroll
        for (int in = 0; in < 4; ++in) {
            v16h bf = load_bf(sB, wn + in * 16 + l15, half);
#pragma unroll
            for (int im = 0; im < 2; ++im)
                acc[im][in] = WMMA_F16(af[im], bf, acc[im][in]);
        }
    }

#pragma unroll
    for (int im = 0; im < 2; ++im)
#pragma unroll
        for (int in = 0; in < 4; ++in) {
            const int n = n0 + wn + in * 16 + l15;
            const float bb = b3[n];
#pragma unroll
            for (int v = 0; v < 8; ++v) {
                const int mloc = wm + im * 16 + v + 8 * half;
                if (m0 + mloc < cnt) {
                    float val = acc[im][in][v] + bb;
                    if (toklist) {
                        int entry = toklist[off + m0 + mloc];
                        int t = entry & 0xffff, slot = (entry >> 16) & 1;
                        val *= gatev[t * 2 + slot];
                        Yout[((size_t)t * 2 + slot) * Ndim + n] = val;
                    } else {
                        Yout[(size_t)(m0 + mloc) * Ndim + n] = val;
                    }
                }
            }
        }
}

// ---- 5) combine: out += ybuf[:,0,:] + ybuf[:,1,:] --------------------------

__global__ __launch_bounds__(256) void combine_kernel(
    float* __restrict__ out, const float* __restrict__ ybuf)
{
    const size_t i = (size_t)blockIdx.x * 256 + threadIdx.x;  // float4 index
    const size_t t = i >> 10;            // DIM/4 = 1024 float4 per row
    const size_t c = i & 1023;
    float4* o = (float4*)out + t * 1024 + c;
    const float4* y0 = (const float4*)ybuf + (t * 2) * 1024 + c;
    const float4* y1 = y0 + 1024;
    float4 a = *o, b = *y0, d = *y1;
    a.x += b.x + d.x; a.y += b.y + d.y; a.z += b.z + d.z; a.w += b.w + d.w;
    *o = a;
}

// ---------------------------------------------------------------------------

extern "C" void kernel_launch(void* const* d_in, const int* in_sizes, int n_in,
                              void* d_out, int out_size, void* d_ws, size_t ws_size,
                              hipStream_t stream) {
    (void)in_sizes; (void)n_in; (void)out_size; (void)ws_size;

    const float* x     = (const float*)d_in[0];
    const float* rw    = (const float*)d_in[1];
    const float* rb    = (const float*)d_in[2];
    const float* rbias = (const float*)d_in[3];
    const float* ew1   = (const float*)d_in[4];
    const float* eb1   = (const float*)d_in[5];
    const float* ew2   = (const float*)d_in[6];
    const float* eb2   = (const float*)d_in[7];
    const float* ew3   = (const float*)d_in[8];
    const float* eb3   = (const float*)d_in[9];
    const float* sw1   = (const float*)d_in[10];
    const float* sb1   = (const float*)d_in[11];
    const float* sw2   = (const float*)d_in[12];
    const float* sb2   = (const float*)d_in[13];
    const float* sw3   = (const float*)d_in[14];
    const float* sb3   = (const float*)d_in[15];
    float* out = (float*)d_out;

    char* ws = (char*)d_ws;
    int*       topidx  = (int*)ws;          ws += (size_t)TOKENS * 2 * sizeof(int);
    float*     topval  = (float*)ws;        ws += (size_t)TOKENS * 2 * sizeof(float);
    int*       offsets = (int*)ws;          ws += 128;
    int*       toklist = (int*)ws;          ws += (size_t)TOKENS * 2 * sizeof(int);
    _Float16*  hbuf    = (_Float16*)ws;     ws += (size_t)TOKENS * 2 * INTER * sizeof(_Float16);
    _Float16*  hsbuf   = (_Float16*)ws;     ws += (size_t)TOKENS * SH_INTER * sizeof(_Float16);
    float*     ybuf    = (float*)ws;        // TOKENS*2*DIM floats

    router_kernel<<<TOKENS / 4, 128, 0, stream>>>(x, rw, rb, rbias, topidx, topval);

    build_lists<<<1, 512, 0, stream>>>(topidx, toklist, offsets);

    fused_gemm_silu<<<dim3(INTER / TN, TOKENS / TM, NEXP), 256, 0, stream>>>(
        x, toklist, offsets, 0, ew1, eb1, ew2, eb2, DIM, INTER, hbuf);

    fused_gemm_silu<<<dim3(SH_INTER / TN, TOKENS / TM, 1), 256, 0, stream>>>(
        x, nullptr, nullptr, TOKENS, sw1, sb1, sw2, sb2, DIM, SH_INTER, hsbuf);

    gemm_out<<<dim3(DIM / TN, TOKENS / TM, 1), 256, 0, stream>>>(
        hsbuf, nullptr, nullptr, TOKENS, sw3, sb3, nullptr, SH_INTER, DIM, out);

    gemm_out<<<dim3(DIM / TN, TOKENS / TM, NEXP), 256, 0, stream>>>(
        hbuf, toklist, offsets, 0, ew3, eb3, topval, INTER, DIM, ybuf);

    combine_kernel<<<(TOKENS * DIM / 4) / 256, 256, 0, stream>>>(out, ybuf);
}
